// LCALayer_33260226740523
// MI455X (gfx1250) — matrix operand order repaired and checked by reading, not compiled, not run on hardware
//
#include <hip/hip_runtime.h>

typedef __attribute__((ext_vector_type(16))) _Float16 v16h;
typedef __attribute__((ext_vector_type(8)))  float    v8f;

#define LAM 0.1f
#define ETA 0.1f

// problem dims
#define BN 64          // batch
#define FD 64          // dictionary atoms / channels
#define HO 56
#define WO 56
// padded f16 activation map (NHWC, channels innermost):
// y index in [-1,56] -> 58 rows, x index in [-1,64] -> 66 cols (compute x padded to 64)
#define APH 58
#define APW 66

#define G_ELEMS (9*2*4*32*16)   // [tap][kchunk][mtile][lane][e] = 36864 halves (72KB)

__device__ __forceinline__ float softthr(float u) {
    return fmaxf(u - LAM, 0.f) - fmaxf(-u - LAM, 0.f);
}

// ---------------------------------------------------------------------------
// Gram kernel: G[f',f,dy,dx] = sum_{c,i,j} d[f',c,i,j] * d[f,c,i-4dy,j-4dx]
// emitted directly in a WMMA A-operand lane layout using contiguous k per lane half (same order as the B loads):
//   element (lane,e): m = mtile*16 + lane%16 ; kbase = lane<16?0:16
//   k_local = kbase + e   (K = kchunk*32 + k_local)
// ---------------------------------------------------------------------------
__global__ void gram_kernel(const float* __restrict__ dict, _Float16* __restrict__ Gsw) {
    int t = blockIdx.x * blockDim.x + threadIdx.x;
    if (t >= G_ELEMS) return;
    int e    =  t        & 15;
    int lane = (t >> 4)  & 31;
    int mt   = (t >> 9)  & 3;
    int kc   = (t >> 11) & 1;
    int tap  =  t >> 12;                 // 0..8
    int dy = tap / 3 - 1, dx = tap % 3 - 1;
    int m = mt * 16 + (lane & 15);
    int klocal = ((lane < 16) ? 0 : 16) + e;
    int k = kc * 32 + klocal;
    float s = 0.f;
    for (int c = 0; c < 3; ++c)
        for (int i = 0; i < 8; ++i) {
            int i2 = i - 4 * dy;
            if (i2 < 0 || i2 >= 8) continue;
            for (int j = 0; j < 8; ++j) {
                int j2 = j - 4 * dx;
                if (j2 < 0 || j2 >= 8) continue;
                s += dict[(m * 3 + c) * 64 + i * 8 + j] *
                     dict[(k * 3 + c) * 64 + i2 * 8 + j2];
            }
        }
    Gsw[t] = (_Float16)s;
}

// ---------------------------------------------------------------------------
// b = conv2d(x, D, stride 4, pad 2), output NHWC fp32 (B,56,56,64)
// one thread per pixel, 64-filter accumulator in registers, D in LDS [pe][f]
// ---------------------------------------------------------------------------
__global__ void bconv_kernel(const float* __restrict__ x, const float* __restrict__ dict,
                             float* __restrict__ b) {
    __shared__ float ldsD[192 * 64];
    for (int idx = threadIdx.x; idx < 192 * 64; idx += blockDim.x) {
        int pe = idx >> 6, f = idx & 63;
        ldsD[idx] = dict[f * 192 + pe];
    }
    __syncthreads();
    int p = blockIdx.x * blockDim.x + threadIdx.x;     // 64*56*56 pixels
    if (p >= BN * HO * WO) return;
    int xo = p % WO, yo = (p / WO) % HO, n = p / (HO * WO);
    float acc[64];
#pragma unroll
    for (int f = 0; f < 64; ++f) acc[f] = 0.f;
    for (int c = 0; c < 3; ++c)
        for (int i = 0; i < 8; ++i) {
            int iy = yo * 4 - 2 + i;
            if (iy < 0 || iy >= 224) continue;
            for (int j = 0; j < 8; ++j) {
                int ix = xo * 4 - 2 + j;
                if (ix < 0 || ix >= 224) continue;
                float xv = x[((size_t)(n * 3 + c) * 224 + iy) * 224 + ix];
                const float* dp = &ldsD[(c * 64 + i * 8 + j) * 64];
#pragma unroll
                for (int f = 0; f < 64; ++f) acc[f] = fmaf(xv, dp[f], acc[f]);
            }
        }
    float* bp = &b[(size_t)p * 64];
#pragma unroll
    for (int f = 0; f < 64; ++f) bp[f] = acc[f];
}

// ---------------------------------------------------------------------------
// One LCA iteration. Per wave: 16-pixel strip, M=64 filters (4 WMMA M-tiles),
// K = 9 taps x 64 channels = 18 chunks of 32 -> 72 v_wmma_f32_16x16x32_f16.
//   inh_plus_a = G * a_prev  (3x3 Gram conv via GEMM)
//   u += 0.1*(b - u - inh_plus_a + soft(u));  a_next = soft(u)
// ---------------------------------------------------------------------------
__global__ void __launch_bounds__(256)
lca_iter_kernel(const _Float16* __restrict__ aprev, _Float16* __restrict__ anext,
                const float* __restrict__ b, float* __restrict__ u,
                const _Float16* __restrict__ Gsw) {
    __shared__ _Float16 ldsG[G_ELEMS];   // 72 KB
    {
        const uint4* src = (const uint4*)Gsw;
        uint4*       dst = (uint4*)ldsG;
        for (int i = threadIdx.x; i < G_ELEMS / 8; i += blockDim.x) dst[i] = src[i];
    }
    __syncthreads();

    int wave = threadIdx.x >> 5;
    int lane = threadIdx.x & 31;
    int task = blockIdx.x * (blockDim.x >> 5) + wave;  // 64*56*4 = 14336 exactly
    int xt = task & 3;
    int y  = (task >> 2) % 56;
    int n  = task / (56 * 4);
    int x0 = xt * 16;
    int px = x0 + (lane & 15);

    v8f acc0 = {}, acc1 = {}, acc2 = {}, acc3 = {};
    int koffB = (lane < 16) ? 0 : 16;
    const _Float16* abase = aprev + (size_t)n * APH * APW * 64;

#pragma unroll
    for (int tap = 0; tap < 9; ++tap) {
        int dy = tap / 3 - 1, dx = tap % 3 - 1;
        int yy = y + dy + 1;
        int xx = px + dx + 1;
        const _Float16* bp = abase + ((size_t)yy * APW + xx) * 64;
        const _Float16* gtap = &ldsG[(size_t)tap * 2 * 2048];
#pragma unroll
        for (int kc = 0; kc < 2; ++kc) {
            v16h bfrag = *(const v16h*)(bp + kc * 32 + koffB);
            const _Float16* gp = gtap + (size_t)kc * 2048 + lane * 16;
            v16h a0 = *(const v16h*)(gp + 0 * 512);
            v16h a1 = *(const v16h*)(gp + 1 * 512);
            v16h a2 = *(const v16h*)(gp + 2 * 512);
            v16h a3 = *(const v16h*)(gp + 3 * 512);
            acc0 = __builtin_amdgcn_wmma_f32_16x16x32_f16(false, a0, false, bfrag, (short)0, acc0, false, false);
            acc1 = __builtin_amdgcn_wmma_f32_16x16x32_f16(false, a1, false, bfrag, (short)0, acc1, false, false);
            acc2 = __builtin_amdgcn_wmma_f32_16x16x32_f16(false, a2, false, bfrag, (short)0, acc2, false, false);
            acc3 = __builtin_amdgcn_wmma_f32_16x16x32_f16(false, a3, false, bfrag, (short)0, acc3, false, false);
        }
    }

    if (px < 56) {
        int voff = (lane < 16) ? 0 : 8;
        size_t pix = ((size_t)n * 56 + y) * 56 + px;
        float*       up  = u + pix * 64;
        const float* bpp = b + pix * 64;
        _Float16*    ap  = anext + (((size_t)n * APH + (y + 1)) * APW + (px + 1)) * 64;
#pragma unroll
        for (int mt = 0; mt < 4; ++mt) {
            v8f acc = (mt == 0) ? acc0 : (mt == 1) ? acc1 : (mt == 2) ? acc2 : acc3;
#pragma unroll
            for (int v = 0; v < 8; ++v) {
                int f = mt * 16 + voff + v;
                float uo = up[f];
                float ao = softthr(uo);
                float un = uo + ETA * (bpp[f] - uo - acc[v] + ao);
                up[f] = un;
                ap[f] = (_Float16)softthr(un);
            }
        }
    }
}

// ---------------------------------------------------------------------------
// final: out (NCHW fp32) = soft(u) with NHWC->NCHW transpose
// ---------------------------------------------------------------------------
__global__ void out_kernel(const float* __restrict__ u, float* __restrict__ out) {
    int t = blockIdx.x * blockDim.x + threadIdx.x;
    if (t >= BN * FD * HO * WO) return;
    int x = t % 56, y = (t / 56) % 56, f = (t / 3136) % 64, n = t / (3136 * 64);
    float uo = u[(((size_t)n * 56 + y) * 56 + x) * 64 + f];
    out[t] = softthr(uo);
}

extern "C" void kernel_launch(void* const* d_in, const int* in_sizes, int n_in,
                              void* d_out, int out_size, void* d_ws, size_t ws_size,
                              hipStream_t stream) {
    const float* x    = (const float*)d_in[0];   // (64,3,224,224)
    const float* dict = (const float*)d_in[1];   // (64,3,8,8), unit-norm atoms
    float* out = (float*)d_out;                  // (64,64,56,56)

    char* ws = (char*)d_ws;
    const size_t U_BYTES = (size_t)BN * HO * WO * FD * 4;        // 51,380,224
    const size_t A_BYTES = (size_t)BN * APH * APW * FD * 2;      // 31,358,976
    float*    u   = (float*)   (ws);
    float*    b   = (float*)   (ws + U_BYTES);
    _Float16* aA  = (_Float16*)(ws + 2 * U_BYTES);
    _Float16* aB  = (_Float16*)(ws + 2 * U_BYTES + A_BYTES);
    _Float16* Gsw = (_Float16*)(ws + 2 * U_BYTES + 2 * A_BYTES);

    // state init (graph-capturable async memsets): u = 0, a halo buffers = 0
    hipMemsetAsync(u,  0, U_BYTES, stream);
    hipMemsetAsync(aA, 0, A_BYTES, stream);
    hipMemsetAsync(aB, 0, A_BYTES, stream);

    gram_kernel<<<G_ELEMS / 256, 256, 0, stream>>>(dict, Gsw);
    bconv_kernel<<<(BN * HO * WO + 255) / 256, 256, 0, stream>>>(x, dict, b);

    const int iterBlocks = (BN * HO * 4) / 8;   // 14336 wave-tasks / 8 waves per block
    for (int it = 0; it < 10; ++it) {
        const _Float16* ap = (it & 1) ? aB : aA;
        _Float16*       an = (it & 1) ? aA : aB;
        lca_iter_kernel<<<iterBlocks, 256, 0, stream>>>(ap, an, b, u, Gsw);
    }

    out_kernel<<<(BN * FD * HO * WO + 255) / 256, 256, 0, stream>>>(u, out);
}
